// VQVAE_78752520339938
// MI455X (gfx1250) — matrix operand order repaired and checked by reading, not compiled, not run on hardware
//
#include <hip/hip_runtime.h>

typedef __attribute__((ext_vector_type(16))) __bf16 v16bf;
typedef __attribute__((ext_vector_type(8)))  float  v8f;

__device__ __forceinline__ unsigned short f2bf(float f) {
  unsigned int u = __float_as_uint(f);
  u = (u + 0x7FFFu + ((u >> 16) & 1u)) >> 16;   // round-to-nearest-even
  return (unsigned short)u;
}

// ---------------- conv1: direct (Cin = 1, tiny) ----------------
__global__ __launch_bounds__(256) void k_conv1(const float* __restrict__ x,
                                               const float* __restrict__ w,
                                               const float* __restrict__ b,
                                               float* __restrict__ out, int Bn) {
  int t = blockIdx.x * blockDim.x + threadIdx.x;
  int total = Bn * 128 * 24 * 24;
  if (t >= total) return;
  int ox = t % 24; int tt = t / 24;
  int oy = tt % 24; tt /= 24;
  int co = tt % 128; int bb = tt / 128;
  float acc = b[co];
  const float* xb = x + (size_t)bb * 784;
  const float* wb = w + co * 25;
#pragma unroll
  for (int ky = 0; ky < 5; ++ky)
#pragma unroll
    for (int kx = 0; kx < 5; ++kx)
      acc = fmaf(xb[(oy + ky) * 28 + (ox + kx)], wb[ky * 5 + kx], acc);
  out[t] = acc > 0.f ? acc : 0.f;
}

// ---------------- weight prep: fp32 -> bf16 WMMA A-fragment layout ----------------
// 16-bit A 16x32 layout (ISA 7.12.2): lane = 16*h + m ; element e -> K:
//   e<8 : K = e + 8*h ;  e>=8 : K = (e-8) + 8*h + 16
// Fragment id fi = ((kpos*4 + ci_chunk)*8 + co_tile); ushort index = fi*512 + lane*16 + e.
__global__ __launch_bounds__(256) void k_prep_w(const float* __restrict__ w,
                                                unsigned short* __restrict__ wf,
                                                int KH, int KW, int deconv) {
  int t = blockIdx.x * blockDim.x + threadIdx.x;
  int total = KH * KW * 16384;
  if (t >= total) return;
  int e    = t & 15;
  int lane = (t >> 4) & 31;
  int wv   = (t >> 9) & 7;
  int cc   = (t >> 12) & 3;
  int kp   = t >> 14;
  int ky = kp / KW, kx = kp - ky * KW;
  int m = lane & 15, h = lane >> 4;
  int K = (e < 8) ? (e + 8 * h) : (e - 8 + 8 * h + 16);
  int co = wv * 16 + m;
  int ci = cc * 32 + K;
  float v;
  if (deconv)  // torch ConvTranspose2d weight (in,out,kH,kW): transpose + flip
    v = w[(((size_t)ci * 128 + co) * KH + (KH - 1 - ky)) * KW + (KW - 1 - kx)];
  else
    v = w[(((size_t)co * 128 + ci) * KH + ky) * KW + kx];
  wf[t] = f2bf(v);
}

// ---------------- implicit-GEMM conv via v_wmma_f32_16x16x32_bf16 ----------------
// Compile-time geometry => no runtime division, fully unrolled stage loop,
// double-buffered LDS B tiles (one barrier per K-step, loads for s+1 overlap
// WMMAs of s). M=128 co (8 waves x 16), N=32 pixels (2 tiles), K = 4 ci-chunks
// x KH*KW taps. STR = lhs-dilation (deconv); fp32 NCHW in/out, f32 accumulate.
template <int IH, int IW, int OH, int OW, int KH, int KW, int PAD, int STR, int RELU>
__global__ __launch_bounds__(256) void k_conv_wmma(
    const float* __restrict__ in, const unsigned short* __restrict__ wf,
    const float* __restrict__ bias, float* __restrict__ out, int Bn) {
  constexpr int NST = KH * KW * 4;       // K-steps
  constexpr int CS  = IH * IW;           // input channel stride
  const int tid  = threadIdx.x;
  const int lane = tid & 31;
  const int wv   = tid >> 5;
  const int n16  = lane & 15;
  const int hh   = lane >> 4;
  const int Npix = Bn * OH * OW;
  const int pixb = blockIdx.x * 32;

  __shared__ v16bf btile[2][64];         // double-buffered pair of 32x16 bf16 tiles

  // staging role: thread t loads pixel sn for channel pair kpr (pixel-fastest)
  const int sn  = tid & 15;
  const int kpr = tid >> 4;              // 0..15 -> (h, e/2)
  const int sh  = kpr >> 3;
  const int sep = kpr & 7;

  const float* ibase[2]; int soy[2], sox[2]; bool spv[2];
#pragma unroll
  for (int j = 0; j < 2; ++j) {
    int p = pixb + j * 16 + sn;
    spv[j] = p < Npix;
    int pp = spv[j] ? p : 0;
    int bb = pp / (OH * OW); int r = pp - bb * (OH * OW);
    soy[j] = r / OW; sox[j] = r - soy[j] * OW;
    ibase[j] = in + (size_t)bb * (128 * CS);
  }

  v8f acc0, acc1;
#pragma unroll
  for (int r = 0; r < 8; ++r) {
    float bv = bias[wv * 16 + r + 8 * hh];
    acc0[r] = bv; acc1[r] = bv;
  }

  auto stage = [&](int s, int buf) {
    const int kp = s >> 2, cc = s & 3;
    const int ky = kp / KW, kx = kp - (kp / KW) * KW;
    const int ci0 = cc * 32 + 2 * sep + 8 * sh + 8 * (sep >> 2);
    unsigned int* bst = (unsigned int*)&btile[buf][0];
#pragma unroll
    for (int j = 0; j < 2; ++j) {
      int dy = soy[j] + ky - PAD, dx = sox[j] + kx - PAD;
      bool ok = spv[j] & (dy >= 0) & (dx >= 0);
      if (STR != 1) ok = ok & ((dy % STR) == 0) & ((dx % STR) == 0);
      int iy = dy / STR, ix = dx / STR;
      ok = ok & (iy < IH) & (ix < IW);
      const float* ip = ibase[j] + (size_t)ci0 * CS + iy * IW + ix;
      float f0 = ok ? ip[0]  : 0.f;
      float f1 = ok ? ip[CS] : 0.f;
      bst[j * 256 + (sh * 16 + sn) * 8 + sep] =
          (unsigned int)f2bf(f0) | ((unsigned int)f2bf(f1) << 16);
    }
  };

  stage(0, 0);
#pragma unroll
  for (int s = 0; s < NST; ++s) {
    __syncthreads();                      // buf[s&1] ready; buf[(s+1)&1] free
    if (s + 1 < NST) stage(s + 1, (s + 1) & 1);
    const v16bf* afp = (const v16bf*)wf + ((size_t)s * 8 + wv) * 32 + lane;
    v16bf afrag = *afp;                   // 32B contiguous per lane, L2-hot
    v16bf b0 = btile[s & 1][lane];        // 32B contiguous ds reads
    v16bf b1 = btile[s & 1][32 + lane];
    acc0 = __builtin_amdgcn_wmma_f32_16x16x32_bf16(false, afrag, false, b0,
                                                   (short)0, acc0, false, false);
    acc1 = __builtin_amdgcn_wmma_f32_16x16x32_bf16(false, afrag, false, b1,
                                                   (short)0, acc1, false, false);
  }

  // C/D layout: lane (h,n), VGPR r -> M = r + 8h, N = n
#pragma unroll
  for (int j = 0; j < 2; ++j) {
    int p = pixb + j * 16 + n16;
    if (p < Npix) {
      int bb = p / (OH * OW); int r = p - bb * (OH * OW);
      int oy = r / OW; int ox = r - oy * OW;
      float* ob = out + (((size_t)bb * 128) * OH + oy) * OW + ox;
      v8f a = j ? acc1 : acc0;
#pragma unroll
      for (int r8 = 0; r8 < 8; ++r8) {
        int co = wv * 16 + r8 + 8 * hh;
        float vv = a[r8];
        if (RELU) vv = vv > 0.f ? vv : 0.f;
        ob[(size_t)co * (OH * OW)] = vv;
      }
    }
  }
}

// ---------------- 2x2 maxpool ----------------
__global__ __launch_bounds__(256) void k_maxpool2(const float* __restrict__ in,
                                                  float* __restrict__ out,
                                                  int N, int OH, int OW) {
  int t = blockIdx.x * blockDim.x + threadIdx.x;
  int total = N * OH * OW;
  if (t >= total) return;
  int ox = t % OW; int tt = t / OW; int oy = tt % OH; int pl = tt / OH;
  const float* ip = in + ((size_t)pl * (2 * OH) + 2 * oy) * (2 * OW) + 2 * ox;
  float m0 = fmaxf(ip[0], ip[1]);
  float m1 = fmaxf(ip[2 * OW], ip[2 * OW + 1]);
  out[t] = fmaxf(m0, m1);
}

// ---------------- VQ: argmax of distance (faithful to reference) ----------------
__global__ __launch_bounds__(128) void k_nn(const float* __restrict__ z,
                                            const float* __restrict__ dict,
                                            int* __restrict__ index) {
  int p = blockIdx.x;
  int t = threadIdx.x;                  // code id 0..127
  int b = p / 36; int r = p - b * 36; int y = r / 6; int x = r - y * 6;
  __shared__ float zs[128];
  __shared__ float dist[128];
  zs[t] = z[(((size_t)b * 128 + t) * 6 + y) * 6 + x];
  __syncthreads();
  float dot = 0.f, wn = 0.f, z2 = 0.f;
  const float* wr = dict + (size_t)t * 128;
  for (int c = 0; c < 128; ++c) {
    float zc = zs[c], wc = wr[c];
    dot = fmaf(zc, wc, dot); wn = fmaf(wc, wc, wn); z2 = fmaf(zc, zc, z2);
  }
  dist[t] = z2 + wn - 2.f * dot;
  __syncthreads();
  if (t == 0) {
    int best = 0; float bd = dist[0];
    for (int i = 1; i < 128; ++i)
      if (dist[i] > bd) { bd = dist[i]; best = i; }   // first-max tie-break
    index[p] = best;
  }
}

__global__ __launch_bounds__(256) void k_gather(const int* __restrict__ index,
                                                const float* __restrict__ dict,
                                                float* __restrict__ val, int Bn) {
  int t = blockIdx.x * blockDim.x + threadIdx.x;
  int total = Bn * 36 * 128;
  if (t >= total) return;
  int c = t & 127; int p = t >> 7;
  int b = p / 36; int r = p - b * 36; int y = r / 6; int x = r - y * 6;
  val[(((size_t)b * 128 + c) * 6 + y) * 6 + x] = dict[(size_t)index[p] * 128 + c];
}

// ---------------- 1x1 conv 128 -> 1 ----------------
__global__ __launch_bounds__(256) void k_mu(const float* __restrict__ f,
                                            const float* __restrict__ mw,
                                            const float* __restrict__ mb,
                                            float* __restrict__ mu, int Bn) {
  int t = blockIdx.x * blockDim.x + threadIdx.x;
  int total = Bn * 784;
  if (t >= total) return;
  int b = t / 784; int r = t - b * 784;
  const float* fb = f + (size_t)b * 128 * 784 + r;
  float acc = mb[0];
  for (int c = 0; c < 128; ++c) acc = fmaf(fb[(size_t)c * 784], mw[c], acc);
  mu[t] = acc;
}

// ---------------- reductions ----------------
__global__ void k_zero2(float* acc) { if (threadIdx.x < 2) acc[threadIdx.x] = 0.f; }

__global__ __launch_bounds__(256) void k_sqdiff(const float* __restrict__ a,
                                                const float* __restrict__ b,
                                                float* __restrict__ slot, int n) {
  __shared__ float s[256];
  float v = 0.f;
  for (int i = blockIdx.x * 256 + threadIdx.x; i < n; i += gridDim.x * 256) {
    float d = a[i] - b[i];
    v = fmaf(d, d, v);
  }
  s[threadIdx.x] = v; __syncthreads();
  for (int off2 = 128; off2 > 0; off2 >>= 1) {
    if (threadIdx.x < off2) s[threadIdx.x] += s[threadIdx.x + off2];
    __syncthreads();
  }
  if (threadIdx.x == 0) atomicAdd(slot, s[0]);
}

__global__ void k_final(const float* __restrict__ acc, const int* __restrict__ index,
                        float* __restrict__ out, int P, int nrec, int nz) {
  int t = blockIdx.x * blockDim.x + threadIdx.x;
  if (t == 0) out[0] = acc[0] / (float)nrec;                 // loss_rec
  else if (t == 1) out[1] = acc[1] / (float)nz * 5.0f;       // dict_loss
  else if (t == 2) out[2] = acc[1] / (float)nz * 1.25f;      // enc_loss
  else if (t == 3) out[3] = 0.f;                             // var_loss
  if (t < P) out[4 + t] = (float)index[t];
}

extern "C" void kernel_launch(void* const* d_in, const int* in_sizes, int n_in,
                              void* d_out, int out_size, void* d_ws, size_t ws_size,
                              hipStream_t stream) {
  const float* x     = (const float*)d_in[0];
  const float* w1    = (const float*)d_in[1];
  const float* b1    = (const float*)d_in[2];
  const float* w2    = (const float*)d_in[3];
  const float* b2    = (const float*)d_in[4];
  const float* w3    = (const float*)d_in[5];
  const float* b3    = (const float*)d_in[6];
  const float* t1w   = (const float*)d_in[7];
  const float* t1b   = (const float*)d_in[8];
  const float* t2w   = (const float*)d_in[9];
  const float* t2b   = (const float*)d_in[10];
  const float* t3w   = (const float*)d_in[11];
  const float* t3b   = (const float*)d_in[12];
  const float* t4w   = (const float*)d_in[13];
  const float* t4b   = (const float*)d_in[14];
  const float* mw    = (const float*)d_in[15];
  const float* mb    = (const float*)d_in[16];
  const float* dictw = (const float*)d_in[17];

  const int Bn = in_sizes[0] / 784;
  const int P  = Bn * 36;

  char* ws = (char*)d_ws;
  size_t off = 0;
  auto alloc = [&](size_t bytes) -> void* {
    void* p = ws + off;
    off = (off + bytes + 255) & ~(size_t)255;
    return p;
  };
  float* R1 = (float*)alloc((size_t)Bn * 128 * 784 * 4);   // ping
  float* R2 = (float*)alloc((size_t)Bn * 128 * 784 * 4);   // pong
  float* Z  = (float*)alloc((size_t)Bn * 128 * 36 * 4);
  float* V  = (float*)alloc((size_t)Bn * 128 * 36 * 4);
  unsigned short* wf2 = (unsigned short*)alloc((size_t)25 * 16384 * 2);
  unsigned short* wf3 = (unsigned short*)alloc((size_t)25 * 16384 * 2);
  unsigned short* wt1 = (unsigned short*)alloc((size_t)16 * 16384 * 2);
  unsigned short* wt2 = (unsigned short*)alloc((size_t)16 * 16384 * 2);
  unsigned short* wt3 = (unsigned short*)alloc((size_t)25 * 16384 * 2);
  unsigned short* wt4 = (unsigned short*)alloc((size_t)25 * 16384 * 2);
  int*   idx = (int*)alloc((size_t)P * 4);
  float* acc = (float*)alloc(256);

  auto cdiv = [](int a, int b) { return (a + b - 1) / b; };

  // weight prep (bf16 fragment layout; deconvs pre-flipped/transposed)
  k_prep_w<<<cdiv(25 * 16384, 256), 256, 0, stream>>>(w2,  wf2, 5, 5, 0);
  k_prep_w<<<cdiv(25 * 16384, 256), 256, 0, stream>>>(w3,  wf3, 5, 5, 0);
  k_prep_w<<<cdiv(16 * 16384, 256), 256, 0, stream>>>(t1w, wt1, 4, 4, 1);
  k_prep_w<<<cdiv(16 * 16384, 256), 256, 0, stream>>>(t2w, wt2, 4, 4, 1);
  k_prep_w<<<cdiv(25 * 16384, 256), 256, 0, stream>>>(t3w, wt3, 5, 5, 1);
  k_prep_w<<<cdiv(25 * 16384, 256), 256, 0, stream>>>(t4w, wt4, 5, 5, 1);

  // encoder
  k_conv1<<<cdiv(Bn * 128 * 24 * 24, 256), 256, 0, stream>>>(x, w1, b1, R1, Bn);
  k_conv_wmma<24,24,24,24,5,5,2,1,1><<<cdiv(Bn * 24 * 24, 32), 256, 0, stream>>>(
      R1, wf2, b2, R2, Bn);
  k_maxpool2<<<cdiv(Bn * 128 * 12 * 12, 256), 256, 0, stream>>>(R2, R1, Bn * 128, 12, 12);
  k_conv_wmma<12,12,12,12,5,5,2,1,0><<<cdiv(Bn * 12 * 12, 32), 256, 0, stream>>>(
      R1, wf3, b3, R2, Bn);
  k_maxpool2<<<cdiv(Bn * 128 * 6 * 6, 256), 256, 0, stream>>>(R2, Z, Bn * 128, 6, 6);

  // VQ
  k_nn<<<P, 128, 0, stream>>>(Z, dictw, idx);
  k_gather<<<cdiv(P * 128, 256), 256, 0, stream>>>(idx, dictw, V, Bn);

  // decoder (deconvs as lhs-dilated convs with pad = k-1)
  k_conv_wmma< 6, 6, 9, 9,4,4,3,1,1><<<cdiv(Bn *  9 *  9, 32), 256, 0, stream>>>(
      V,  wt1, t1b, R1, Bn);
  k_conv_wmma< 9, 9,20,20,4,4,3,2,1><<<cdiv(Bn * 20 * 20, 32), 256, 0, stream>>>(
      R1, wt2, t2b, R2, Bn);
  k_conv_wmma<20,20,24,24,5,5,4,1,1><<<cdiv(Bn * 24 * 24, 32), 256, 0, stream>>>(
      R2, wt3, t3b, R1, Bn);
  k_conv_wmma<24,24,28,28,5,5,4,1,1><<<cdiv(Bn * 28 * 28, 32), 256, 0, stream>>>(
      R1, wt4, t4b, R2, Bn);
  k_mu<<<cdiv(Bn * 784, 256), 256, 0, stream>>>(R2, mw, mb, R1, Bn);

  // losses (acc re-zeroed every call -> deterministic under graph replay)
  k_zero2<<<1, 32, 0, stream>>>(acc);
  k_sqdiff<<<512, 256, 0, stream>>>(R1, x, acc + 0, Bn * 784);
  k_sqdiff<<<512, 256, 0, stream>>>(V,  Z, acc + 1, Bn * 128 * 36);
  k_final<<<cdiv(P + 4, 256), 256, 0, stream>>>(acc, idx, (float*)d_out,
      P, Bn * 784, Bn * 128 * 36);
}
